// MambaLayer_33354716021198
// MI455X (gfx1250) — matrix operand order repaired and checked by reading, not compiled, not run on hardware
//
#include <hip/hip_runtime.h>
#include <hip/hip_bf16.h>

// ---------------------------------------------------------------------------
// Types
// ---------------------------------------------------------------------------
typedef __attribute__((ext_vector_type(16))) __bf16 v16bf;
typedef __attribute__((ext_vector_type(8)))  float  v8f;
typedef __attribute__((ext_vector_type(4)))  unsigned u32x4;

__device__ __forceinline__ unsigned short f32_to_bf16(float f) {
    unsigned u = __builtin_bit_cast(unsigned, f);
    unsigned r = u + 0x7FFFu + ((u >> 16) & 1u);
    return (unsigned short)(r >> 16);
}
__device__ __forceinline__ float bf16_to_f32(unsigned short h) {
    unsigned u = ((unsigned)h) << 16;
    return __builtin_bit_cast(float, u);
}

// CDNA5 async global->LDS copy (per-lane 16B; tracked by ASYNCcnt).
__device__ __forceinline__ void async_copy_b128(const void* gptr, void* lds_ptr) {
    unsigned lds_off = (unsigned)(uintptr_t)lds_ptr;       // flat->LDS: low 32 bits
    unsigned long long ga = (unsigned long long)(uintptr_t)gptr;
    asm volatile("global_load_async_to_lds_b128 %0, %1, off"
                 :
                 : "v"(lds_off), "v"(ga)
                 : "memory");
}
__device__ __forceinline__ void wait_async0() {
#if __has_builtin(__builtin_amdgcn_s_wait_asynccnt)
    __builtin_amdgcn_s_wait_asynccnt(0);
#else
    asm volatile("s_wait_asynccnt 0x0" ::: "memory");
#endif
}

// ---------------------------------------------------------------------------
// Problem constants (B=2, L=2048, d_model=1024, d_inner=2048, d_state=16)
// ---------------------------------------------------------------------------
#define BSZ 2
#define LSEQ 2048
#define DM 1024
#define DI 2048
#define DSTATE 16
#define MROWS (BSZ * LSEQ)   // 4096

// ---------------------------------------------------------------------------
// fp32 -> bf16 conversion
// ---------------------------------------------------------------------------
__global__ __launch_bounds__(256) void cvt_bf16_kernel(const float* __restrict__ in,
                                                       unsigned short* __restrict__ out, int n) {
    int i = blockIdx.x * 256 + threadIdx.x;
    if (i < n) out[i] = f32_to_bf16(in[i]);
}

// ---------------------------------------------------------------------------
// WMMA bf16 GEMM:  C[M,N] = A[M,K] * W[N,K]^T (+bias), A/W row-major bf16.
// Block tile 128x128, 8 waves (4x2); each wave 32x64 via 2x4 WMMA frags.
// K staged 64 at a time, DOUBLE-BUFFERED via global_load_async_to_lds_b128.
// LDS row stride 72 elements (144B) -> conflict-free frag reads on 64 banks.
// ---------------------------------------------------------------------------
#define GBM 128
#define GBN 128
#define GBK 64
#define SAPAD 72

__device__ __forceinline__ v16bf load_fragA(const unsigned short* sA, int mrow0, int kk, int lane) {
    int m = mrow0 + (lane & 15);
    int h = lane >> 4;
    union { v16bf v; unsigned u[8]; } f;
    const unsigned short* p = &sA[m * SAPAD];
#pragma unroll
    for (int j = 0; j < 4; ++j)
        f.u[j] = *reinterpret_cast<const unsigned*>(&p[kk + h * 8 + 2 * j]);
#pragma unroll
    for (int j = 0; j < 4; ++j)
        f.u[4 + j] = *reinterpret_cast<const unsigned*>(&p[kk + 16 + h * 8 + 2 * j]);
    return f.v;
}
__device__ __forceinline__ v16bf load_fragB(const unsigned short* sB, int nrow0, int kk, int lane) {
    int n = nrow0 + (lane & 15);
    int h = lane >> 4;
    union { v16bf v; unsigned u[8]; } f;
    const unsigned short* p = &sB[n * SAPAD + kk + h * 8];
#pragma unroll
    for (int j = 0; j < 8; ++j)
        f.u[j] = *reinterpret_cast<const unsigned*>(&p[2 * j + (j >= 4 ? 8 : 0)]);
    return f.v;
}

template <bool OUT_BF16>
__global__ __launch_bounds__(256) void gemm_bf16_kernel(const unsigned short* __restrict__ A,
                                                        const unsigned short* __restrict__ W,
                                                        void* __restrict__ Cout,
                                                        const float* __restrict__ bias,
                                                        int M, int N, int K) {
    __shared__ unsigned short sA[2][GBM * SAPAD];
    __shared__ unsigned short sB[2][GBN * SAPAD];

    const int m0 = blockIdx.y * GBM;
    const int n0 = blockIdx.x * GBN;
    const int lane = threadIdx.x & 31;
    const int wave = threadIdx.x >> 5;
    const int wm = wave >> 1;   // 0..3 -> 32 rows each
    const int wn = wave & 1;    // 0..1 -> 64 cols each

    // Staging map: 1024 16B-chunks per 128x64 tile, 4 chunks per thread.
    int chRow[4], chCol[4];
#pragma unroll
    for (int i = 0; i < 4; ++i) {
        int chunk = threadIdx.x + i * 256;
        chRow[i] = chunk >> 3;
        chCol[i] = (chunk & 7) << 3;
    }

    v8f acc[2][4] = {};

    // prologue: stage tile 0 into buffer 0
#pragma unroll
    for (int i = 0; i < 4; ++i) {
        async_copy_b128(&A[(size_t)(m0 + chRow[i]) * K + chCol[i]],
                        &sA[0][chRow[i] * SAPAD + chCol[i]]);
        async_copy_b128(&W[(size_t)(n0 + chRow[i]) * K + chCol[i]],
                        &sB[0][chRow[i] * SAPAD + chCol[i]]);
    }
    wait_async0();
    __syncthreads();

    int p = 0;
    for (int k0 = 0; k0 < K; k0 += GBK) {
        // issue async loads for next K tile into the other buffer
        if (k0 + GBK < K) {
            int kn = k0 + GBK;
#pragma unroll
            for (int i = 0; i < 4; ++i) {
                async_copy_b128(&A[(size_t)(m0 + chRow[i]) * K + kn + chCol[i]],
                                &sA[p ^ 1][chRow[i] * SAPAD + chCol[i]]);
                async_copy_b128(&W[(size_t)(n0 + chRow[i]) * K + kn + chCol[i]],
                                &sB[p ^ 1][chRow[i] * SAPAD + chCol[i]]);
            }
        }
        // compute on current buffer
#pragma unroll
        for (int kk = 0; kk < GBK; kk += 32) {
            v16bf a0 = load_fragA(sA[p], wm * 32 + 0,  kk, lane);
            v16bf a1 = load_fragA(sA[p], wm * 32 + 16, kk, lane);
            v16bf b0 = load_fragB(sB[p], wn * 64 + 0,  kk, lane);
            v16bf b1 = load_fragB(sB[p], wn * 64 + 16, kk, lane);
            v16bf b2 = load_fragB(sB[p], wn * 64 + 32, kk, lane);
            v16bf b3 = load_fragB(sB[p], wn * 64 + 48, kk, lane);
            acc[0][0] = __builtin_amdgcn_wmma_f32_16x16x32_bf16(false, a0, false, b0, (short)0, acc[0][0], false, false);
            acc[0][1] = __builtin_amdgcn_wmma_f32_16x16x32_bf16(false, a0, false, b1, (short)0, acc[0][1], false, false);
            acc[0][2] = __builtin_amdgcn_wmma_f32_16x16x32_bf16(false, a0, false, b2, (short)0, acc[0][2], false, false);
            acc[0][3] = __builtin_amdgcn_wmma_f32_16x16x32_bf16(false, a0, false, b3, (short)0, acc[0][3], false, false);
            acc[1][0] = __builtin_amdgcn_wmma_f32_16x16x32_bf16(false, a1, false, b0, (short)0, acc[1][0], false, false);
            acc[1][1] = __builtin_amdgcn_wmma_f32_16x16x32_bf16(false, a1, false, b1, (short)0, acc[1][1], false, false);
            acc[1][2] = __builtin_amdgcn_wmma_f32_16x16x32_bf16(false, a1, false, b2, (short)0, acc[1][2], false, false);
            acc[1][3] = __builtin_amdgcn_wmma_f32_16x16x32_bf16(false, a1, false, b3, (short)0, acc[1][3], false, false);
        }
        // next-tile async loads complete + everyone done reading current buffer
        wait_async0();
        __syncthreads();
        p ^= 1;
    }

    // D layout: lane n = lane&15, half = lane>>4; VGPR r -> M = r + 8*half
    const int h = lane >> 4;
    const int nl = lane & 15;
#pragma unroll
    for (int i = 0; i < 2; ++i)
#pragma unroll
        for (int j = 0; j < 4; ++j)
#pragma unroll
            for (int r = 0; r < 8; ++r) {
                int row = m0 + wm * 32 + i * 16 + r + 8 * h;
                int col = n0 + wn * 64 + j * 16 + nl;
                float v = acc[i][j][r];
                if (bias) v += bias[col];
                if (OUT_BF16)
                    ((unsigned short*)Cout)[(size_t)row * N + col] = f32_to_bf16(v);
                else
                    ((float*)Cout)[(size_t)row * N + col] = v;
            }
}

// ---------------------------------------------------------------------------
// Depthwise causal conv (D_CONV=4) + bias + SiLU.  Reads x_p = xr[:, 0:DI].
// ---------------------------------------------------------------------------
__global__ __launch_bounds__(256) void conv_silu_kernel(const unsigned short* __restrict__ xr,
                                                        const float* __restrict__ convw,
                                                        const float* __restrict__ convb,
                                                        unsigned short* __restrict__ xc) {
    int idx = blockIdx.x * 256 + threadIdx.x;
    if (idx >= MROWS * DI) return;
    int c = idx & (DI - 1);
    int l = (idx >> 11) & (LSEQ - 1);
    int b = idx >> 22;
    float acc = convb[c];
    const float* w = &convw[c * 4];
#pragma unroll
    for (int j = 0; j < 4; ++j) {
        int lj = l - 3 + j;
        if (lj >= 0)
            acc += w[j] * bf16_to_f32(xr[((size_t)(b * LSEQ + lj)) * (2 * DI) + c]);
    }
    float s = acc / (1.f + __expf(-acc));  // SiLU
    xc[idx] = f32_to_bf16(s);
}

// ---------------------------------------------------------------------------
// x_proj: BC[m, 0:32] = xc[m,:] @ x_proj_w[32, DI]^T   (tiny N, plain reduce)
// ---------------------------------------------------------------------------
__global__ __launch_bounds__(256) void xproj_kernel(const unsigned short* __restrict__ xc,
                                                    const float* __restrict__ Wx,
                                                    float* __restrict__ BC) {
    int m = blockIdx.x;
    int s = threadIdx.x & 31;
    int part = threadIdx.x >> 5;   // 0..7, each covers 256 of K=2048
    float sum = 0.f;
    const size_t base = (size_t)m * DI;
    for (int k = part * 256; k < part * 256 + 256; ++k)
        sum += bf16_to_f32(xc[base + k]) * Wx[s * DI + k];
    __shared__ float red[8][32];
    red[part][s] = sum;
    __syncthreads();
    if (threadIdx.x < 32) {
        float t = 0.f;
#pragma unroll
        for (int p = 0; p < 8; ++p) t += red[p][threadIdx.x];
        BC[(size_t)m * 32 + threadIdx.x] = t;
    }
}

// ---------------------------------------------------------------------------
// Selective scan, one lane per (channel, state); 16-lane shfl_xor reduction
// for y_t. Fused SiLU(res) gating; writes bf16 for the out_proj GEMM.
// Block = 256 threads = 16 channels x 16 states. 128-step LDS chunking.
// ---------------------------------------------------------------------------
#define TCH 128
__global__ __launch_bounds__(256) void scan_kernel(const unsigned short* __restrict__ xc,
                                                   const float* __restrict__ dtb,
                                                   const float* __restrict__ BC,
                                                   const unsigned short* __restrict__ xr,
                                                   const float* __restrict__ A_log,
                                                   const float* __restrict__ Dp,
                                                   unsigned short* __restrict__ ybf) {
    const int grp = threadIdx.x >> 4;      // local channel 0..15
    const int s   = threadIdx.x & 15;      // state 0..15
    const int g   = blockIdx.x * 16 + grp; // global channel 0..4095
    const int b   = g >> 11;
    const int c   = g & (DI - 1);
    const int c0  = (blockIdx.x * 16) & (DI - 1);

    const float Al = A_log[s];
    const float Dc = Dp[c];
    float h = 0.f;

    __shared__ float          sBC[TCH][32];
    __shared__ unsigned short sU[TCH][16];
    __shared__ float          sDT[TCH][16];

    for (int t0 = 0; t0 < LSEQ; t0 += TCH) {
        __syncthreads();
        for (int i = threadIdx.x; i < TCH * 32; i += 256)
            sBC[i >> 5][i & 31] = BC[(((size_t)(b * LSEQ + t0)) << 5) + i];
        for (int i = threadIdx.x; i < TCH * 16; i += 256) {
            int tt = i >> 4, cc = i & 15;
            size_t off = ((size_t)(b * LSEQ + t0 + tt)) * DI + c0 + cc;
            sU[tt][cc]  = xc[off];
            sDT[tt][cc] = dtb[off];
        }
        __syncthreads();
        for (int tt = 0; tt < TCH; ++tt) {
            float u   = bf16_to_f32(sU[tt][grp]);
            float dtv = sDT[tt][grp];
            float dA  = __expf(dtv * Al);
            float Bt  = sBC[tt][s];
            float Ct  = sBC[tt][16 + s];
            h = dA * h + dtv * Bt * u;
            float v = h * Ct;
            v += __shfl_xor(v, 1, 32);
            v += __shfl_xor(v, 2, 32);
            v += __shfl_xor(v, 4, 32);
            v += __shfl_xor(v, 8, 32);
            if (s == 0) {
                float y = v + Dc * u;
                size_t row = (size_t)(b * LSEQ + t0 + tt);
                float res = bf16_to_f32(xr[row * (2 * DI) + DI + c]);
                float gate = res / (1.f + __expf(-res));
                ybf[row * DI + c] = f32_to_bf16(y * gate);
            }
        }
    }
}

// ---------------------------------------------------------------------------
// Residual add + LayerNorm over d_model=1024. One block per row.
// ---------------------------------------------------------------------------
__global__ __launch_bounds__(256) void ln_kernel(const float* __restrict__ outp,
                                                 const float* __restrict__ x,
                                                 const float* __restrict__ gamma,
                                                 const float* __restrict__ beta,
                                                 float* __restrict__ dout) {
    const int row = blockIdx.x;
    const size_t base = (size_t)row * DM;
    float z[4];
    float sm = 0.f;
#pragma unroll
    for (int i = 0; i < 4; ++i) {
        int col = threadIdx.x + i * 256;
        z[i] = outp[base + col] + x[base + col];
        sm += z[i];
    }
    __shared__ float red[8];
    __shared__ float bcast;
    for (int m = 16; m > 0; m >>= 1) sm += __shfl_xor(sm, m, 32);
    if ((threadIdx.x & 31) == 0) red[threadIdx.x >> 5] = sm;
    __syncthreads();
    if (threadIdx.x == 0) {
        float t = 0.f;
#pragma unroll
        for (int w = 0; w < 8; ++w) t += red[w];
        bcast = t * (1.f / (float)DM);
    }
    __syncthreads();
    const float mu = bcast;
    float vs = 0.f;
#pragma unroll
    for (int i = 0; i < 4; ++i) { float d = z[i] - mu; vs += d * d; }
    for (int m = 16; m > 0; m >>= 1) vs += __shfl_xor(vs, m, 32);
    __syncthreads();
    if ((threadIdx.x & 31) == 0) red[threadIdx.x >> 5] = vs;
    __syncthreads();
    if (threadIdx.x == 0) {
        float t = 0.f;
#pragma unroll
        for (int w = 0; w < 8; ++w) t += red[w];
        bcast = t * (1.f / (float)DM);
    }
    __syncthreads();
    const float rstd = rsqrtf(bcast + 1e-5f);
#pragma unroll
    for (int i = 0; i < 4; ++i) {
        int col = threadIdx.x + i * 256;
        dout[base + col] = gamma[col] * (z[i] - mu) * rstd + beta[col];
    }
}

// ---------------------------------------------------------------------------
// Launcher
// ---------------------------------------------------------------------------
extern "C" void kernel_launch(void* const* d_in, const int* in_sizes, int n_in,
                              void* d_out, int out_size, void* d_ws, size_t ws_size,
                              hipStream_t stream) {
    (void)in_sizes; (void)n_in; (void)out_size; (void)ws_size;

    const float* x        = (const float*)d_in[0];   // [2,2048,1024]
    const float* in_projw = (const float*)d_in[1];   // [4096,1024]
    const float* conv_w   = (const float*)d_in[2];   // [2048,1,4]
    const float* conv_b   = (const float*)d_in[3];   // [2048]
    const float* x_projw  = (const float*)d_in[4];   // [32,2048]
    const float* dt_projw = (const float*)d_in[5];   // [2048,2048]
    const float* dt_projb = (const float*)d_in[6];   // [2048]
    const float* A_log    = (const float*)d_in[7];   // [16]
    const float* Dvec     = (const float*)d_in[8];   // [2048]
    const float* out_projw= (const float*)d_in[9];   // [1024,2048]
    const float* ln_gamma = (const float*)d_in[10];  // [1024]
    const float* ln_beta  = (const float*)d_in[11];  // [1024]
    float* dout = (float*)d_out;

    char* ws = (char*)d_ws;
    size_t off = 0;
    auto alloc = [&](size_t bytes) -> void* {
        void* p = ws + off;
        off += (bytes + 255) & ~(size_t)255;
        return p;
    };

    unsigned short* xbf   = (unsigned short*)alloc((size_t)MROWS * DM * 2);
    unsigned short* W1bf  = (unsigned short*)alloc((size_t)(2 * DI) * DM * 2);
    unsigned short* Wdtbf = (unsigned short*)alloc((size_t)DI * DI * 2);
    unsigned short* Wobf  = (unsigned short*)alloc((size_t)DM * DI * 2);
    unsigned short* xrbf  = (unsigned short*)alloc((size_t)MROWS * 2 * DI * 2);
    unsigned short* xcbf  = (unsigned short*)alloc((size_t)MROWS * DI * 2);
    float*          dtf   = (float*)alloc((size_t)MROWS * DI * 4);
    float*          BC    = (float*)alloc((size_t)MROWS * 32 * 4);
    unsigned short* ybf   = (unsigned short*)alloc((size_t)MROWS * DI * 2);
    float*          outf  = (float*)alloc((size_t)MROWS * DM * 4);

    // 1) downconvert activations + GEMM weights to bf16
    {
        int n0 = MROWS * DM;           // x
        int n1 = 2 * DI * DM;          // in_proj_w
        int n2 = DI * DI;              // dt_proj_w
        int n3 = DM * DI;              // out_proj_w
        cvt_bf16_kernel<<<(n0 + 255) / 256, 256, 0, stream>>>(x, xbf, n0);
        cvt_bf16_kernel<<<(n1 + 255) / 256, 256, 0, stream>>>(in_projw, W1bf, n1);
        cvt_bf16_kernel<<<(n2 + 255) / 256, 256, 0, stream>>>(dt_projw, Wdtbf, n2);
        cvt_bf16_kernel<<<(n3 + 255) / 256, 256, 0, stream>>>(out_projw, Wobf, n3);
    }

    // 2) in_proj GEMM: xr[4096,4096] = x[4096,1024] @ W1[4096,1024]^T (bf16 out)
    gemm_bf16_kernel<true><<<dim3((2 * DI) / GBN, MROWS / GBM), 256, 0, stream>>>(
        xbf, W1bf, xrbf, nullptr, MROWS, 2 * DI, DM);

    // 3) depthwise conv + SiLU -> xc (bf16)
    conv_silu_kernel<<<(MROWS * DI) / 256, 256, 0, stream>>>(xrbf, conv_w, conv_b, xcbf);

    // 4) dt_proj GEMM: dt[4096,2048] = xc @ Wdt^T + dt_proj_b (f32 out)
    gemm_bf16_kernel<false><<<dim3(DI / GBN, MROWS / GBM), 256, 0, stream>>>(
        xcbf, Wdtbf, dtf, dt_projb, MROWS, DI, DI);

    // 5) x_proj: BC[4096,32] = xc @ x_proj_w^T
    xproj_kernel<<<MROWS, 256, 0, stream>>>(xcbf, x_projw, BC);

    // 6) selective scan + fused SiLU(res) gate -> ybf (bf16)
    scan_kernel<<<(BSZ * DI) / 16, 256, 0, stream>>>(
        xcbf, dtf, BC, xrbf, A_log, Dvec, ybf);

    // 7) out_proj GEMM: out[4096,1024] = y @ Wout^T (f32 out)
    gemm_bf16_kernel<false><<<dim3(DM / GBN, MROWS / GBM), 256, 0, stream>>>(
        ybf, Wobf, outf, nullptr, MROWS, DM, DI);

    // 8) residual + LayerNorm
    ln_kernel<<<MROWS, 256, 0, stream>>>(outf, x, ln_gamma, ln_beta, dout);
}
